// EdgeConv_33277406609579
// MI455X (gfx1250) — matrix operand order repaired and checked by reading, not compiled, not run on hardware
//
#include <hip/hip_runtime.h>
#include <hip/hip_bf16.h>
#include <stdint.h>

// ---------------- CDNA5 WMMA types ----------------
typedef _Float16 v16h __attribute__((ext_vector_type(16)));
typedef _Float16 half8 __attribute__((ext_vector_type(8)));
typedef float    v8f  __attribute__((ext_vector_type(8)));

// ---------------- problem sizes ----------------
#define B_    64
#define N_    1024
#define F_    7
#define C_    64
#define KNB   16
#define ROWS  (B_ * N_ * KNB)   // 1,048,576 rows of the (rows x 64) activation tensors
#define TILES (ROWS / 16)       // 65,536 WMMA row-tiles

// LDS tile staging: 16 rows x 64 f16, rows padded to 72 elems (144 B = 36 dwords,
// gcd(36,64)=4 -> the 16 per-lane b128 A-fragment reads hit disjoint bank quads)
#define TROW  72

// ---------------- workspace layout (bytes) ----------------
static constexpr size_t KNN_OFF   = 0;                                   // int[ROWS]          4 MB
static constexpr size_t HA_OFF    = (size_t)4 << 20;                     // f16[ROWS*64]     128 MB
static constexpr size_t HB_OFF    = HA_OFF + (size_t)ROWS * C_ * 2;      // f16[ROWS*64]     128 MB
static constexpr size_t STATS_OFF = HB_OFF + (size_t)ROWS * C_ * 2;      // f32[3*256]
static constexpr size_t PACKW_OFF = STATS_OFF + 3 * 256 * 4;             // u32[1024+2048+2048]

// =====================================================================
// prep: zero stats, pack W0/W1/W2 into WMMA B-fragment layout.
// B-matrix (32x16 f16, 8 VGPRs): lanes 0-15 -> N=lane, VGPR j holds K=2j,2j+1 (j<4), 2j+8,2j+9 (j>=4)
//                                lanes 16-31 -> N=lane-16, K=8+2j,8+2j+1 (j<4), 16+2j,16+2j+1 (j>=4)
// Entry index e = ((kb*4 + nb)*32 + lane)*8 + j
// =====================================================================
__global__ __launch_bounds__(256) void prep_kernel(const float* __restrict__ W0,
                                                   const float* __restrict__ W1,
                                                   const float* __restrict__ W2,
                                                   float* __restrict__ stats,
                                                   uint32_t* __restrict__ packw) {
    int tid = threadIdx.x;
    for (int t = tid; t < 3 * 256; t += 256) stats[t] = 0.0f;   // re-zeroed every launch

    for (int layer = 0; layer < 3; ++layer) {
        const float* W = (layer == 0) ? W0 : ((layer == 1) ? W1 : W2);
        int cin   = (layer == 0) ? 14 : 64;      // layer0 K padded 14 -> 32 with zeros
        int kbcnt = (layer == 0) ? 1 : 2;
        uint32_t* out = packw + ((layer == 0) ? 0 : ((layer == 1) ? 1024 : 3072));
        int total = kbcnt * 4 * 32 * 8;
        for (int e = tid; e < total; e += 256) {
            int j    = e & 7;
            int lane = (e >> 3) & 31;
            int nb   = (e >> 8) & 3;
            int kb   = e >> 10;
            int col  = nb * 16 + (lane & 15);
            int k0   = kb * 32 + ((lane & 16) ? 8 : 0) + 2 * j + ((j >= 4) ? 8 : 0);
            float w0 = (k0     < cin) ? W[(size_t)k0 * 64 + col]       : 0.0f;
            float w1 = (k0 + 1 < cin) ? W[(size_t)(k0 + 1) * 64 + col] : 0.0f;
            union { _Float16 h[2]; uint32_t u; } cv;
            cv.h[0] = (_Float16)w0;
            cv.h[1] = (_Float16)w1;
            out[e] = cv.u;
        }
    }
}

// =====================================================================
// kNN: 4 blocks per batch; x[b] (1024x7 f32 = 28 KB) LDS-resident.
// Each thread owns one point, keeps a register top-16 via replace-max
// (fully unrolled -> stays in VGPRs).
// =====================================================================
__global__ __launch_bounds__(256) void knn_kernel(const float* __restrict__ x,
                                                  int* __restrict__ knn) {
    __shared__ float sx[N_ * F_];
    int b     = blockIdx.x >> 2;
    int chunk = blockIdx.x & 3;
    const float* xb = x + (size_t)b * N_ * F_;
    for (int t = threadIdx.x; t < N_ * F_; t += 256) sx[t] = xb[t];
    __syncthreads();

    int i = chunk * 256 + threadIdx.x;
    float xi[7];
#pragma unroll
    for (int t = 0; t < 7; ++t) xi[t] = sx[i * 7 + t];

    float bd[16]; int bi[16];
#pragma unroll
    for (int p = 0; p < 16; ++p) { bd[p] = 3.0e38f; bi[p] = 0; }
    float worst = 3.0e38f;

    for (int j = 0; j < N_; ++j) {
        float d = 0.0f;
#pragma unroll
        for (int t = 0; t < 7; ++t) { float df = xi[t] - sx[j * 7 + t]; d += df * df; }
        if (d < worst) {
            int pos = 0; float mv = bd[0];
#pragma unroll
            for (int p = 1; p < 16; ++p) if (bd[p] > mv) { mv = bd[p]; pos = p; }
#pragma unroll
            for (int p = 0; p < 16; ++p) if (p == pos) { bd[p] = d; bi[p] = j; }
            worst = bd[0];
#pragma unroll
            for (int p = 1; p < 16; ++p) worst = fmaxf(worst, bd[p]);
        }
    }
#pragma unroll
    for (int p = 0; p < 16; ++p) knn[(size_t)(b * N_ + i) * 16 + p] = bi[p];
}

// =====================================================================
// Layer 0: edge features (14 channels, K padded to 32) x W0 -> h0 (pre-BN, f16)
// + per-channel sum / sumsq accumulation. Output staged through LDS so the
// global store is 4 coalesced b128 per lane (tile is contiguous in memory).
// =====================================================================
__global__ __launch_bounds__(256) void layer0_kernel(const float* __restrict__ x,
                                                     const int* __restrict__ knn,
                                                     const uint32_t* __restrict__ fragB,
                                                     _Float16* __restrict__ h0,
                                                     float* __restrict__ stats0) {
    __shared__ float ssum[64], ssq[64];
    __shared__ _Float16 stile[8][16 * TROW];
    int tid = threadIdx.x;
    if (tid < 64) { ssum[tid] = 0.0f; ssq[tid] = 0.0f; }
    __syncthreads();

    int lane = tid & 31, wave = tid >> 5;
    bool up  = (lane & 16) != 0;
    int tile = blockIdx.x * 8 + wave;
    int row  = tile * 16 + (lane & 15);
    int b    = row >> 14;           // / (1024*16)
    int n    = (row >> 4) & 1023;
    int jn   = knn[row];
    const float* xip = x + (size_t)(b * N_ + n)  * 7;
    const float* xjp = x + (size_t)(b * N_ + jn) * 7;

    float xiv[7], xjv[7];
#pragma unroll
    for (int t = 0; t < 7; ++t) { xiv[t] = xip[t]; xjv[t] = xjp[t]; }

    // A-matrix (16x32 f16): lanes 0-15 -> row M=lane, elems 0-7 = K0..7, 8-15 = K16..23
    //                       lanes 16-31 -> row M=lane-16, K8..15 and K24..31
    // Edge feature c: c<7 -> x_i[c]; 7<=c<14 -> x_j[c-7]-x_i[c-7]; c>=14 -> 0
    v16h A;
#pragma unroll
    for (int t = 0; t < 16; ++t) {
        float v = 0.0f;
        if (!up) {
            if (t < 7)       v = xiv[t];
            else if (t == 7) v = xjv[0] - xiv[0];
        } else {
            if (t < 6)       v = xjv[t + 1] - xiv[t + 1];
        }
        A[t] = (_Float16)v;
    }

    _Float16* st = stile[wave];
#pragma unroll
    for (int nb = 0; nb < 4; ++nb) {
        v16h Bf = *(const v16h*)(fragB + ((size_t)nb * 32 + lane) * 8);
        v8f Cv = {};
        Cv = __builtin_amdgcn_wmma_f32_16x16x32_f16(false, A, false, Bf,
                                                    (short)0, Cv, false, false);
        int cc = nb * 16 + (lane & 15);
        int rl = up ? 8 : 0;
        float s = 0.0f, q = 0.0f;
#pragma unroll
        for (int p = 0; p < 8; ++p) {
            float v = Cv[p];
            st[(rl + p) * TROW + cc] = (_Float16)v;
            s += v; q += v * v;
        }
        atomicAdd(&ssum[cc], s);
        atomicAdd(&ssq[cc], q);
    }
    __syncthreads();

    // coalesced write-back: tile is contiguous 1024 f16 in memory
    _Float16* gdst = h0 + (size_t)tile * 1024;
#pragma unroll
    for (int i = 0; i < 4; ++i) {
        int g = lane * 8 + i * 256;                      // global elem offset in tile
        *(half8*)(gdst + g) = *(const half8*)(st + (g >> 6) * TROW + (g & 63));
    }
    if (tid < 64) {
        atomicAdd(&stats0[tid],      ssum[tid]);
        atomicAdd(&stats0[64 + tid], ssq[tid]);
    }
}

// =====================================================================
// BN finalize: sums -> per-channel scale/shift. Bias terms cancel inside
// BN (constant shift of h shifts mean identically), so they're omitted.
// =====================================================================
__global__ void bn_finalize_kernel(const float* __restrict__ gamma,
                                   const float* __restrict__ beta,
                                   float* __restrict__ stats) {
    int c = threadIdx.x;
    if (c < 64) {
        const float inv_cnt = 1.0f / (float)ROWS;
        float mean = stats[c] * inv_cnt;
        float var  = stats[64 + c] * inv_cnt - mean * mean;
        float r    = rsqrtf(var + 1e-3f);
        float a    = gamma[c] * r;
        stats[128 + c] = a;                   // scale
        stats[192 + c] = beta[c] - mean * a;  // shift
    }
}

// =====================================================================
// Layers 1/2: coalesced b128 load of the contiguous 16x64 f16 tile into LDS,
// per-lane A-fragment reads from LDS, apply prev-layer BN+ReLU in f32,
// GEMM (K=64 -> 2 WMMAs per col-tile, 4 col-tiles), stage pre-BN output in
// LDS, coalesced b128 write-back, accumulate this layer's stats.
// =====================================================================
__global__ __launch_bounds__(256) void layer_gemm_kernel(const _Float16* __restrict__ hIn,
                                                         _Float16* __restrict__ hOut,
                                                         const uint32_t* __restrict__ fragB,
                                                         const float* __restrict__ coefA,
                                                         const float* __restrict__ coefD,
                                                         float* __restrict__ statsOut) {
    __shared__ float ssum[64], ssq[64], sca[64], scd[64];
    __shared__ _Float16 stile[8][16 * TROW];
    int tid = threadIdx.x;
    if (tid < 64) {
        ssum[tid] = 0.0f; ssq[tid] = 0.0f;
        sca[tid] = coefA[tid]; scd[tid] = coefD[tid];
    }
    __syncthreads();

    int lane = tid & 31, wave = tid >> 5;
    bool up   = (lane & 16) != 0;
    int tile  = blockIdx.x * 8 + wave;
    int khalf = up ? 8 : 0;
    _Float16* st = stile[wave];

    // coalesced tile load -> LDS (padded rows)
    const _Float16* gsrc = hIn + (size_t)tile * 1024;
#pragma unroll
    for (int i = 0; i < 4; ++i) {
        int g = lane * 8 + i * 256;
        *(half8*)(st + (g >> 6) * TROW + (g & 63)) = *(const half8*)(gsrc + g);
    }
    __syncthreads();

    // A-fragments from LDS, with BN + ReLU applied in f32
    int r = lane & 15;
    v16h A[2];
#pragma unroll
    for (int kb = 0; kb < 2; ++kb) {
        const _Float16* p = st + r * TROW + kb * 32 + khalf;
        half8 lo = *(const half8*)p;           // K = cb .. cb+7
        half8 hi = *(const half8*)(p + 16);    // K = cb+16 .. cb+23
        int cb = kb * 32 + khalf;
#pragma unroll
        for (int i = 0; i < 8; ++i) {
            A[kb][i]     = (_Float16)fmaxf(sca[cb + i]      * (float)lo[i] + scd[cb + i],      0.0f);
            A[kb][8 + i] = (_Float16)fmaxf(sca[cb + 16 + i] * (float)hi[i] + scd[cb + 16 + i], 0.0f);
        }
    }

    // GEMM + stage output tile in LDS (A-loads of all lanes complete before the
    // wave-wide WMMA issues; LDS ops from one wave execute in order)
#pragma unroll
    for (int nb = 0; nb < 4; ++nb) {
        v8f Cv = {};
#pragma unroll
        for (int kb = 0; kb < 2; ++kb) {
            v16h Bf = *(const v16h*)(fragB + ((size_t)(kb * 4 + nb) * 32 + lane) * 8);
            Cv = __builtin_amdgcn_wmma_f32_16x16x32_f16(false, A[kb], false, Bf,
                                                        (short)0, Cv, false, false);
        }
        int cc = nb * 16 + (lane & 15);
        int rl = up ? 8 : 0;
        float s = 0.0f, q = 0.0f;
#pragma unroll
        for (int p = 0; p < 8; ++p) {
            float v = Cv[p];
            st[(rl + p) * TROW + cc] = (_Float16)v;
            s += v; q += v * v;
        }
        atomicAdd(&ssum[cc], s);
        atomicAdd(&ssq[cc], q);
    }
    __syncthreads();

    // coalesced write-back
    _Float16* gdst = hOut + (size_t)tile * 1024;
#pragma unroll
    for (int i = 0; i < 4; ++i) {
        int g = lane * 8 + i * 256;
        *(half8*)(gdst + g) = *(const half8*)(st + (g >> 6) * TROW + (g & 63));
    }
    if (tid < 64) {
        atomicAdd(&statsOut[tid],      ssum[tid]);
        atomicAdd(&statsOut[64 + tid], ssq[tid]);
    }
}

// =====================================================================
// Final: apply BN2+ReLU, mean over 16 neighbor rows, write f32 (B,N,64).
// Loads are coalesced (consecutive c across threads for each k).
// =====================================================================
__global__ __launch_bounds__(256) void final_mean_kernel(const _Float16* __restrict__ h2,
                                                         const float* __restrict__ coefA,
                                                         const float* __restrict__ coefD,
                                                         float* __restrict__ out) {
    int tid = threadIdx.x;
    int c  = tid & 63;
    int bn = blockIdx.x * 4 + (tid >> 6);
    float a = coefA[c], d = coefD[c];
    const _Float16* p = h2 + (size_t)bn * 16 * 64 + c;
    float s = 0.0f;
#pragma unroll
    for (int k = 0; k < 16; ++k) s += fmaxf(a * (float)p[k * 64] + d, 0.0f);
    out[(size_t)bn * 64 + c] = s * (1.0f / 16.0f);
}

// =====================================================================
extern "C" void kernel_launch(void* const* d_in, const int* in_sizes, int n_in,
                              void* d_out, int out_size, void* d_ws, size_t ws_size,
                              hipStream_t stream) {
    const float* x   = (const float*)d_in[0];
    const float* W0  = (const float*)d_in[1];
    // d_in[2] = b0 (unused: cancels in BN)
    const float* g0  = (const float*)d_in[3];
    const float* be0 = (const float*)d_in[4];
    const float* W1  = (const float*)d_in[5];
    const float* g1  = (const float*)d_in[7];
    const float* be1 = (const float*)d_in[8];
    const float* W2  = (const float*)d_in[9];
    const float* g2  = (const float*)d_in[11];
    const float* be2 = (const float*)d_in[12];

    char* ws = (char*)d_ws;
    int*       knn   = (int*)(ws + KNN_OFF);
    _Float16*  hA    = (_Float16*)(ws + HA_OFF);
    _Float16*  hB    = (_Float16*)(ws + HB_OFF);
    float*     stats = (float*)(ws + STATS_OFF);     // 3 x 256 floats
    uint32_t*  packw = (uint32_t*)(ws + PACKW_OFF);
    float* st0 = stats;
    float* st1 = stats + 256;
    float* st2 = stats + 512;

    prep_kernel<<<1, 256, 0, stream>>>(W0, W1, W2, stats, packw);
    knn_kernel<<<B_ * 4, 256, 0, stream>>>(x, knn);

    layer0_kernel<<<TILES / 8, 256, 0, stream>>>(x, knn, packw, hA, st0);
    bn_finalize_kernel<<<1, 64, 0, stream>>>(g0, be0, st0);

    layer_gemm_kernel<<<TILES / 8, 256, 0, stream>>>(hA, hB, packw + 1024,
                                                     st0 + 128, st0 + 192, st1);
    bn_finalize_kernel<<<1, 64, 0, stream>>>(g1, be1, st1);

    layer_gemm_kernel<<<TILES / 8, 256, 0, stream>>>(hB, hA, packw + 3072,
                                                     st1 + 128, st1 + 192, st2);
    bn_finalize_kernel<<<1, 64, 0, stream>>>(g2, be2, st2);

    final_mean_kernel<<<B_ * N_ / 4, 256, 0, stream>>>(hA, st2 + 128, st2 + 192,
                                                       (float*)d_out);
}
